// Attention_38646115729848
// MI455X (gfx1250) — compile-verified, matrix operand order checked by reading, not run on hardware
//
#include <hip/hip_runtime.h>
#include <hip/hip_bf16.h>
#include <stdint.h>

// ---------------- WMMA / vector types ----------------
typedef __attribute__((ext_vector_type(16))) _Float16 v16h;
typedef __attribute__((ext_vector_type(8)))  float    v8f;
typedef __attribute__((ext_vector_type(4)))  int      v4i;

union Frag16 { v16h v; unsigned int u[8]; };
union HU { _Float16 h; unsigned short u; };

#define BB 2048
#define TT 200
#define FS 128

// Async global->LDS copy path (gfx1250). Guarded so compile never breaks.
#if defined(__HIP_DEVICE_COMPILE__) && __has_builtin(__builtin_amdgcn_global_load_async_to_lds_b128)
#define USE_ASYNC_LDS 1
typedef __attribute__((address_space(1))) v4i* as1_v4i_p;   // global v4i*
typedef __attribute__((address_space(3))) v4i* as3_v4i_p;   // LDS v4i*
#else
#define USE_ASYNC_LDS 0
#endif

// Dynamic LDS layout (bytes)
#define F_STRIDE   132                       // padded row stride (floats)
#define F_BYTES    (208 * F_STRIDE * 4)      // 109824  facts f32 [208][132]
#define H_OFF      F_BYTES                   // h1 per-wave [8][16][100] f16
#define H_BYTES    (8 * 16 * 100 * 2)        // 25600
#define QSH_OFF    (H_OFF + H_BYTES)         // 135424
#define CSH_OFF    (QSH_OFF + 128 * 4)       // 135936
#define SBUF_OFF   (CSH_OFF + 80 * 4)        // 136256
#define RED_OFF    (SBUF_OFF + 208 * 4)      // 137088
#define SMEM_BYTES (RED_OFF + 256 * 4)       // 138112

// =====================================================================
// Kernel 1: q = PReLU(query @ Wq^T + bq); c_b = b1 + q @ (W1a + W1c)^T
// =====================================================================
__global__ void __launch_bounds__(128) prep_q_c(
    const float* __restrict__ query, const float* __restrict__ Wq,
    const float* __restrict__ bq, const float* __restrict__ aprelu,
    const float* __restrict__ W1, const float* __restrict__ b1,
    float* __restrict__ qws, float* __restrict__ cws) {
  __shared__ float qs[128];
  __shared__ float qproj[128];
  const int b = blockIdx.x, j = threadIdx.x;
  qs[j] = query[b * 128 + j];
  __syncthreads();
  const float* wrow = Wq + j * 128;
  float acc = bq[j];
#pragma unroll 4
  for (int k = 0; k < 128; ++k) acc = fmaf(qs[k], wrow[k], acc);
  const float a = aprelu[0];
  const float qv = acc > 0.0f ? acc : a * acc;
  qws[b * 128 + j] = qv;
  qproj[j] = qv;
  __syncthreads();
  if (j < 80) {
    const float* w = W1 + j * 512;
    float c = b1[j];
#pragma unroll 4
    for (int k = 0; k < 128; ++k) c = fmaf(qproj[k], w[k] + w[256 + k], c);
    cws[b * 80 + j] = c;
  }
}

// =====================================================================
// Kernel 2: pack Wc=[W1b-W1c | W1d] (K=256,N=80) and padded W2 (K=96,N=48)
// into f16 WMMA B-fragment layout.
// =====================================================================
__global__ void __launch_bounds__(256) pack_weights(
    const float* __restrict__ W1, const float* __restrict__ W2,
    unsigned int* __restrict__ WcB, unsigned int* __restrict__ W2B) {
  const int stride = gridDim.x * blockDim.x;
  const int tid = blockIdx.x * blockDim.x + threadIdx.x;

  for (int idx = tid; idx < 5 * 8 * 32 * 8; idx += stride) {
    const int i    = idx & 7;
    const int lane = (idx >> 3) & 31;
    const int kk   = (idx >> 8) & 7;
    const int jt   = idx >> 11;
    const int n    = jt * 16 + (lane & 15);
    const int kb   = kk * 32 + ((i & 4) ? 16 : 0) + (lane >> 4) * 8 + (i & 3) * 2;
    unsigned int pack = 0;
#pragma unroll
    for (int p = 0; p < 2; ++p) {
      const int k = kb + p;
      float v;
      if (k < 128) v = W1[n * 512 + 128 + k] - W1[n * 512 + 256 + k];
      else         v = W1[n * 512 + 384 + (k - 128)];
      HU hu; hu.h = (_Float16)v;
      pack |= ((unsigned int)hu.u) << (16 * p);
    }
    WcB[idx] = pack;
  }

  for (int idx = tid; idx < 3 * 3 * 32 * 8; idx += stride) {
    const int i    = idx & 7;
    const int lane = (idx >> 3) & 31;
    const int kk   = (idx >> 8) % 3;
    const int jt   = (idx >> 8) / 3;
    const int n    = jt * 16 + (lane & 15);
    const int kb   = kk * 32 + ((i & 4) ? 16 : 0) + (lane >> 4) * 8 + (i & 3) * 2;
    unsigned int pack = 0;
#pragma unroll
    for (int p = 0; p < 2; ++p) {
      const int k = kb + p;
      const float v = (n < 40 && k < 80) ? W2[n * 80 + k] : 0.0f;
      HU hu; hu.h = (_Float16)v;
      pack |= ((unsigned int)hu.u) << (16 * p);
    }
    W2B[idx] = pack;
  }
}

// =====================================================================
// Kernel 3 (fused): one block = one batch. facts staged once in LDS,
// WMMA GEMM1 + sigmoid + WMMA GEMM2 + W3 dot -> logits, block softmax,
// scale-out from LDS copy. 256 threads = 8 waves; 13 M-tiles of 16 rows.
// =====================================================================
__global__ void __launch_bounds__(256) din_fused(
    const float* __restrict__ facts, const float* __restrict__ qws,
    const float* __restrict__ cws, const unsigned int* __restrict__ WcB,
    const unsigned int* __restrict__ W2B, const float* __restrict__ b2,
    const float* __restrict__ W3, const float* __restrict__ b3,
    const unsigned char* __restrict__ mask,
    float* __restrict__ out, float* __restrict__ scores_out) {
  extern __shared__ char smem[];
  float*    F    = (float*)smem;                 // [208][F_STRIDE] facts f32
  _Float16* H    = (_Float16*)(smem + H_OFF);    // [8 waves][16][100] h1 f16
  float*    qsh  = (float*)(smem + QSH_OFF);     // [128]
  float*    csh  = (float*)(smem + CSH_OFF);     // [80]
  float*    sbuf = (float*)(smem + SBUF_OFF);    // [208] logits -> scores
  float*    red  = (float*)(smem + RED_OFF);     // [256]

  const int b    = blockIdx.x;
  const int tid  = threadIdx.x;                  // 0..255
  const int lane = tid & 31;
  const int wv   = tid >> 5;                     // 0..7
  const int half = lane >> 4;
  const int lm   = lane & 15;

  if (tid < 128)      qsh[tid] = qws[b * 128 + tid];
  else if (tid < 208) csh[tid - 128] = cws[b * 80 + (tid - 128)];

  // ---- stage facts (f32) into LDS once ----
  const float* fb = facts + (size_t)b * TT * FS;
#if USE_ASYNC_LDS
  for (int e = tid; e < TT * 32; e += 256) {     // 6400 x b128, 25/thread
    const int t = e >> 5, c4 = e & 31;
    __builtin_amdgcn_global_load_async_to_lds_b128(
        (as1_v4i_p)(fb + (size_t)e * 4),
        (as3_v4i_p)(F + t * F_STRIDE + c4 * 4),
        0, 0);
  }
#else
  for (int e = tid; e < TT * 32; e += 256) {
    const int t = e >> 5, c4 = e & 31;
    const float4 f = *(const float4*)(fb + (size_t)e * 4);
    *(float4*)(F + t * F_STRIDE + c4 * 4) = f;
  }
#endif
  // zero pad rows 200..207
  for (int e = tid; e < 8 * 32; e += 256) {
    const int t = 200 + (e >> 5), c4 = e & 31;
    float4 z; z.x = 0.f; z.y = 0.f; z.z = 0.f; z.w = 0.f;
    *(float4*)(F + t * F_STRIDE + c4 * 4) = z;
  }
#if USE_ASYNC_LDS
  asm volatile("s_wait_asynccnt 0x0" ::: "memory");
#endif
  __syncthreads();

  const float b3v = b3[0];
  _Float16* Hw = H + wv * 16 * 100;              // this wave's h1 buffer

  for (int mt = wv; mt < 13; mt += 8) {          // waves 0..4 do 2 tiles
    __builtin_amdgcn_wave_barrier();
    const int row = mt * 16 + lm;                // A row for this lane
    const float* Frow = F + row * F_STRIDE;

    // Build A fragments [f | q*f] directly from LDS (branch is compile-time
    // per kk after unroll: kk<4 -> f part, kk>=4 -> q*f part).
    Frag16 af[8];
#pragma unroll
    for (int kk = 0; kk < 8; ++kk) {
#pragma unroll
      for (int i = 0; i < 8; ++i) {
        const int k = kk * 32 + ((i & 4) ? 16 : 0) + half * 8 + (i & 3) * 2;
        float x0, x1;
        if (k < 128) { x0 = Frow[k]; x1 = Frow[k + 1]; }
        else {
          const int kq = k - 128;
          x0 = Frow[kq] * qsh[kq];
          x1 = Frow[kq + 1] * qsh[kq + 1];
        }
        HU h0, h1; h0.h = (_Float16)x0; h1.h = (_Float16)x1;
        af[kk].u[i] = (unsigned int)h0.u | ((unsigned int)h1.u << 16);
      }
    }

    // zero H pad cols 80..95 (wave-private buffer)
    for (int e = lane; e < 16 * 16; e += 32)
      Hw[(e >> 4) * 100 + 80 + (e & 15)] = (_Float16)0.0f;

    // ---- GEMM1: [16 x 256] @ [256 x 80] ----
    for (int n = 0; n < 5; ++n) {
      v8f acc = {};
#pragma unroll
      for (int kk = 0; kk < 8; ++kk) {
        Frag16 bm;
        const uint4* bp = (const uint4*)(WcB + ((((n * 8 + kk) * 32) + lane) << 3));
        const uint4 p0 = bp[0], p1 = bp[1];
        bm.u[0] = p0.x; bm.u[1] = p0.y; bm.u[2] = p0.z; bm.u[3] = p0.w;
        bm.u[4] = p1.x; bm.u[5] = p1.y; bm.u[6] = p1.z; bm.u[7] = p1.w;
        acc = __builtin_amdgcn_wmma_f32_16x16x32_f16(
            false, af[kk].v, false, bm.v, (short)0, acc, false, false);
      }
      const int col = n * 16 + lm;
      const float bias = csh[col];
#pragma unroll
      for (int r = 0; r < 8; ++r) {
        const float x = acc[r] + bias;
        const float hv = 1.0f / (1.0f + __expf(-x));
        Hw[(r + 8 * half) * 100 + col] = (_Float16)hv;   // row m = r + 8*half
      }
    }

    __builtin_amdgcn_wave_barrier();   // LDS in-order per wave; block reorder

    // ---- GEMM2: [16 x 96] @ [96 x 48] + sigmoid + W3 dot ----
    float part[8];
#pragma unroll
    for (int r = 0; r < 8; ++r) part[r] = 0.0f;

    for (int n2 = 0; n2 < 3; ++n2) {
      v8f acc = {};
#pragma unroll
      for (int kk = 0; kk < 3; ++kk) {
        Frag16 a, bm;
#pragma unroll
        for (int i = 0; i < 8; ++i) {
          const int k = kk * 32 + ((i & 4) ? 16 : 0) + half * 8 + (i & 3) * 2;
          a.u[i] = *(const unsigned int*)&Hw[lm * 100 + k];
        }
        const uint4* bp = (const uint4*)(W2B + ((((n2 * 3 + kk) * 32) + lane) << 3));
        const uint4 p0 = bp[0], p1 = bp[1];
        bm.u[0] = p0.x; bm.u[1] = p0.y; bm.u[2] = p0.z; bm.u[3] = p0.w;
        bm.u[4] = p1.x; bm.u[5] = p1.y; bm.u[6] = p1.z; bm.u[7] = p1.w;
        acc = __builtin_amdgcn_wmma_f32_16x16x32_f16(
            false, a.v, false, bm.v, (short)0, acc, false, false);
      }
      const int col = n2 * 16 + lm;                 // valid < 40
      const float bias = (col < 40) ? b2[col] : 0.0f;
      const float w3v  = (col < 40) ? W3[col] : 0.0f;
#pragma unroll
      for (int r = 0; r < 8; ++r) {
        const float hv = 1.0f / (1.0f + __expf(-(acc[r] + bias)));
        part[r] = fmaf(hv, w3v, part[r]);
      }
    }

    // row-sum across the 16 lanes of each half -> logits
#pragma unroll
    for (int r = 0; r < 8; ++r) {
      float v = part[r];
#pragma unroll
      for (int off = 1; off < 16; off <<= 1) v += __shfl_xor(v, off, 32);
      if (lm == 0) sbuf[mt * 16 + r + 8 * half] = v + b3v;
    }
  }
  __syncthreads();

  // ---- masked softmax over T (block-wide) ----
  const float NEGBIG = -4294967295.0f;             // -(2^32 - 1)
  float sm = NEGBIG;
  if (tid < TT) sm = mask[b * TT + tid] ? sbuf[tid] : NEGBIG;
  red[tid] = sm;
  __syncthreads();
  for (int s = 128; s > 0; s >>= 1) {
    if (tid < s) red[tid] = fmaxf(red[tid], red[tid + s]);
    __syncthreads();
  }
  const float mx = red[0];
  __syncthreads();
  float ev = 0.0f;
  if (tid < TT) ev = __expf(sm - mx);
  red[tid] = ev;
  __syncthreads();
  for (int s = 128; s > 0; s >>= 1) {
    if (tid < s) red[tid] += red[tid + s];
    __syncthreads();
  }
  const float inv = 1.0f / red[0];
  __syncthreads();
  if (tid < TT) {
    const float sc = ev * inv;
    sbuf[tid] = sc;
    scores_out[b * TT + tid] = sc;
  }
  __syncthreads();

  // ---- output = facts * score, straight from the LDS facts copy ----
  float4* ob = (float4*)(out + (size_t)b * TT * FS);
  for (int e = tid; e < TT * 32; e += 256) {
    const int t = e >> 5, c4 = e & 31;
    const float sc = sbuf[t];
    float4 f = *(const float4*)(F + t * F_STRIDE + c4 * 4);
    f.x *= sc; f.y *= sc; f.z *= sc; f.w *= sc;
    ob[e] = f;
  }
}

// =====================================================================
extern "C" void kernel_launch(void* const* d_in, const int* in_sizes, int n_in,
                              void* d_out, int out_size, void* d_ws, size_t ws_size,
                              hipStream_t stream) {
  const float*         query = (const float*)d_in[0];
  const float*         facts = (const float*)d_in[1];
  const unsigned char* mask  = (const unsigned char*)d_in[2];  // numpy bool = 1 byte
  const float*         Wq    = (const float*)d_in[3];
  const float*         bq    = (const float*)d_in[4];
  const float*         ap    = (const float*)d_in[5];
  const float*         W1    = (const float*)d_in[6];
  const float*         b1    = (const float*)d_in[7];
  const float*         W2    = (const float*)d_in[8];
  const float*         b2    = (const float*)d_in[9];
  const float*         W3    = (const float*)d_in[10];
  const float*         b3    = (const float*)d_in[11];

  float* out    = (float*)d_out;
  float* scores = out + (size_t)BB * TT * FS;

  // workspace layout (~1.5 MB)
  float* ws  = (float*)d_ws;
  float* qws = ws;                       // B*128
  float* cws = qws + BB * 128;           // B*80
  unsigned int* WcB = (unsigned int*)(cws + BB * 80);  // 10240 dwords
  unsigned int* W2B = WcB + 10240;                     // 2304 dwords

  // allow >64KB dynamic LDS (gfx1250 WGP has 320KB)
  (void)hipFuncSetAttribute((const void*)din_fused,
                            hipFuncAttributeMaxDynamicSharedMemorySize,
                            SMEM_BYTES);

  prep_q_c<<<BB, 128, 0, stream>>>(query, Wq, bq, ap, W1, b1, qws, cws);
  pack_weights<<<40, 256, 0, stream>>>(W1, W2, WcB, W2B);
  din_fused<<<BB, 256, SMEM_BYTES, stream>>>(
      facts, qws, cws, WcB, W2B, b2, W3, b3, mask, out, scores);
}
